// PerformerModel_86552180948992
// MI455X (gfx1250) — matrix operand order repaired and checked by reading, not compiled, not run on hardware
//
#include <hip/hip_runtime.h>
#include <hip/hip_bf16.h>

// ---------------- model dims ----------------
#define BB   2
#define SS   2048
#define VV   32000
#define EE   512
#define HH   8
#define DHH  64
#define MFF  256
#define LL   4
#define ROWS 4096              // B*S
#define HM   2048              // H*MFF  (feature layout [ROWS, H*M])
#define DA   80                // augmented attention width: 64 v-cols + 1 ones + 15 zero
#define NCH  16                // S / 128 chunks

#define DN_SCALE 0.35355339059f   // 64^-0.25
#define RATIO    0.0625f          // 256^-0.5

typedef __bf16 bf16;
typedef __attribute__((ext_vector_type(16))) __bf16 v16bf;
typedef __attribute__((ext_vector_type(8)))  float  v8f;
typedef __attribute__((ext_vector_type(4)))  unsigned int u32x4;
typedef __attribute__((ext_vector_type(8)))  int i32x8;
typedef __attribute__((ext_vector_type(4)))  int i32x4;

#if defined(__has_builtin)
#if __has_builtin(__builtin_amdgcn_tensor_load_to_lds) && __has_builtin(__builtin_amdgcn_s_wait_tensorcnt)
#define HAVE_TDM 1
#else
#define HAVE_TDM 0
#endif
#else
#define HAVE_TDM 0
#endif

// ---------------- small helpers ----------------
__device__ __forceinline__ bf16 f2bf(float f) {
    unsigned int u = __float_as_uint(f);
    unsigned int r = (u + 0x7fffu + ((u >> 16) & 1u)) >> 16;
    unsigned short s = (unsigned short)r;
    return __builtin_bit_cast(bf16, s);
}

__device__ __forceinline__ v8f wmma_bf16(v16bf a, v16bf b, v8f c) {
    return __builtin_amdgcn_wmma_f32_16x16x32_bf16(
        /*neg_a=*/false, a, /*neg_b=*/false, b,
        /*c_mod=*/(short)0, c, /*reuse_a=*/false, /*reuse_b=*/false);
}

__device__ __forceinline__ float gelu_exact(float x) {
    return 0.5f * x * (1.0f + erff(x * 0.70710678118f));
}

// CDNA5 async copy: GLOBAL_LOAD_ASYNC_TO_LDS_B128 (per-lane 16B global -> 16B LDS, ASYNCcnt)
__device__ __forceinline__ void async_b128(unsigned lds_addr, const void* gptr) {
    asm volatile("global_load_async_to_lds_b128 %0, %1, off"
                 :: "v"(lds_addr), "v"((unsigned long long)gptr)
                 : "memory");
}
__device__ __forceinline__ void wait_async0() {
    asm volatile("s_wait_asynccnt 0" ::: "memory");
}
__device__ __forceinline__ unsigned lds_off(const void* p) {
    return (unsigned)(unsigned long long)p;   // generic LDS ptr: low 32 bits = LDS byte address
}

#if HAVE_TDM
// Tensor Data Mover: 2D tile (tile_d0 elems/row x tile_d1 rows, 2-byte elems) global -> LDS.
// D# per CDNA5 ISA §8: group0 {count=1, lds_addr, global_addr[56:0], type=2},
// group1 {data_size=1(2B), tensor dims, tile dims, dim0 stride}; groups 2/3 zero (2D).
// This toolchain's builtin takes 6 args: (g0 u32x4, g1 i32x8, g2 i32x4, g3 i32x4, i32x8, cpol).
__device__ __forceinline__ void tdm_load_2d(unsigned lds_addr, const void* gaddr,
                                            unsigned tensor_d0, unsigned tensor_d1,
                                            unsigned tile_d0, unsigned tile_d1,
                                            unsigned long long stride0) {
    unsigned long long ga = (unsigned long long)gaddr;
    u32x4 g0;
    g0[0] = 1u;                                               // count=1, user mode
    g0[1] = lds_addr;                                         // lds_addr[31:0]
    g0[2] = (unsigned)ga;                                     // global_addr[31:0]
    g0[3] = (unsigned)((ga >> 32) & 0x01FFFFFFu) | 0x80000000u; // addr[56:32] | type=2
    i32x8 g1;
    g1[0] = 0x00010000;                                       // data_size=1 (2 bytes)
    g1[1] = (int)((tensor_d0 & 0xFFFFu) << 16);               // tensor_dim0[15:0] @bit48
    g1[2] = (int)((tensor_d0 >> 16) | ((tensor_d1 & 0xFFFFu) << 16));
    g1[3] = (int)((tensor_d1 >> 16) | (tile_d0 << 16));       // tile_dim0 @bit112
    g1[4] = (int)(tile_d1 & 0xFFFFu);                         // tile_dim1 (tile_dim2=0)
    g1[5] = (int)(unsigned)(stride0 & 0xFFFFFFFFu);           // tensor_dim0_stride[31:0]
    g1[6] = (int)(unsigned)((stride0 >> 32) & 0xFFFFu);       // stride hi (dim1_stride=0)
    g1[7] = 0;
    i32x4 z4 = {0, 0, 0, 0};
    i32x8 z8 = {0, 0, 0, 0, 0, 0, 0, 0};
    __builtin_amdgcn_tensor_load_to_lds(g0, g1, z4, z4, z8, 0);
}
__device__ __forceinline__ void wait_tensor0() { __builtin_amdgcn_s_wait_tensorcnt(0); }
#endif

// A fragment (16x32 bf16) from row-major storage [rows, ld].
// Lanes 0-15 -> M=lane, K in {0..7,16..23}; lanes 16-31 -> M=lane-16, K in {8..15,24..31}.
__device__ __forceinline__ v16bf load_afrag_rm(const bf16* base, int ld, int r0, int k0, int lane) {
    int m = r0 + (lane & 15);
    int kbase = (lane < 16) ? 0 : 8;
    const bf16* p = base + (long)m * ld + k0;
    v16bf a;
#pragma unroll
    for (int i = 0; i < 8; ++i) {
        int kk = kbase + 2 * i + ((i >= 4) ? 8 : 0);
        a[2 * i]     = p[kk];
        a[2 * i + 1] = p[kk + 1];
    }
    return a;
}

// A fragment (16x32) when storage is k-major [K][ld]: element A[m][k] = base[k*ld + m].
__device__ __forceinline__ v16bf load_afrag_cm(const bf16* base, int ld, int r0, int k0, int lane) {
    int m = r0 + (lane & 15);
    int kbase = (lane < 16) ? 0 : 8;
    v16bf a;
#pragma unroll
    for (int i = 0; i < 8; ++i) {
        int kk = k0 + kbase + 2 * i + ((i >= 4) ? 8 : 0);
        a[2 * i]     = base[(long)kk * ld + m];
        a[2 * i + 1] = base[(long)(kk + 1) * ld + m];
    }
    return a;
}

// B fragment (32x16 bf16) from n-major storage (B^T row-major): element B[k][n] = baseT[n*ld + k].
__device__ __forceinline__ v16bf load_bfrag_nm(const bf16* baseT, int ld, int n0, int k0, int lane) {
    int n = n0 + (lane & 15);
    int koff = (lane < 16) ? 0 : 8;
    const bf16* p = baseT + (long)n * ld + k0 + koff;
    v16bf b;
#pragma unroll
    for (int i = 0; i < 16; ++i) b[i] = p[(i < 8) ? i : i + 8];
    return b;
}

// ---------------- conversion kernels ----------------
__global__ __launch_bounds__(256) void k_f2b(const float* __restrict__ in, bf16* __restrict__ out, long n) {
    long i = (long)blockIdx.x * blockDim.x + threadIdx.x;
    long st = (long)gridDim.x * blockDim.x;
    for (; i < n; i += st) out[i] = f2bf(in[i]);
}

// transpose-convert: in fp32 [K,N] (z-batched) -> out bf16 [N,K], tiled through LDS
__global__ __launch_bounds__(256) void k_f2bT(const float* __restrict__ in, bf16* __restrict__ out,
                                              int K, int N) {
    __shared__ float tile[64][65];
    long mb = (long)blockIdx.z * K * N;
    int k0 = blockIdx.x * 64, n0 = blockIdx.y * 64;
    int tid = threadIdx.x;
    for (int i = tid; i < 4096; i += 256) {
        int r = i >> 6, c = i & 63;
        tile[r][c] = in[mb + (long)(k0 + r) * N + n0 + c];
    }
    __syncthreads();
    for (int i = tid; i < 4096; i += 256) {
        int r = i >> 6, c = i & 63;
        out[mb + (long)(n0 + r) * K + k0 + c] = f2bf(tile[c][r]);
    }
}

// ---------------- embedding + posenc ----------------
__global__ __launch_bounds__(256) void k_embed(const int* __restrict__ toks, const float* __restrict__ emb,
                                               float* __restrict__ out) {
    int row = blockIdx.x;
    int s = row & (SS - 1);
    int tid = threadIdx.x;
    int tok = toks[row];
    float div = expf(-0.035977892f * (float)tid);  // exp(-(2*i)*ln(10000)/512)
    float ang = (float)s * div;
    float sv = sinf(ang), cv = cosf(ang);
    long o = (long)row * EE + tid * 2;
    long e = (long)tok * EE + tid * 2;
    out[o]     = emb[e]     + sv;
    out[o + 1] = emb[e + 1] + cv;
}

// ---------------- layernorm (E=512) ----------------
__global__ __launch_bounds__(256) void k_ln(const float* __restrict__ in, const float* __restrict__ g,
                                            const float* __restrict__ bb, float* __restrict__ outf,
                                            bf16* __restrict__ outb) {
    __shared__ float s1[256], s2[256];
    int row = blockIdx.x, tid = threadIdx.x;
    const float* pr = in + (long)row * EE;
    float a = pr[tid * 2], c = pr[tid * 2 + 1];
    s1[tid] = a + c;
    s2[tid] = a * a + c * c;
    __syncthreads();
    for (int st = 128; st > 0; st >>= 1) {
        if (tid < st) { s1[tid] += s1[tid + st]; s2[tid] += s2[tid + st]; }
        __syncthreads();
    }
    float mu = s1[0] * (1.0f / EE);
    float var = s2[0] * (1.0f / EE) - mu * mu;
    float rstd = rsqrtf(var + 1e-5f);
    int c0 = tid * 2;
    float y0 = (a - mu) * rstd * g[c0] + bb[c0];
    float y1 = (c - mu) * rstd * g[c0 + 1] + bb[c0 + 1];
    long o = (long)row * EE + c0;
    if (outf) { outf[o] = y0; outf[o + 1] = y1; }
    if (outb) { outb[o] = f2bf(y0); outb[o + 1] = f2bf(y1); }
}

// ---------------- generic WMMA GEMM ----------------
// C[M,N] = act(A[M,K]bf16 @ B + bias) + resid ; B given TRANSPOSED: BwT bf16 [N,K].
// 256 threads = 8 waves as 4(m)x2(n); block tile 128x128; wave: 2 m-tiles x 4 n-tiles.
// K-loop double-buffered with GLOBAL_LOAD_ASYNC_TO_LDS_B128.
__global__ __launch_bounds__(256) void k_gemm(
    const bf16* __restrict__ A, int lda, long aBatch,
    const bf16* __restrict__ BwT, int ldbT,
    int M, int N, int K,
    float* __restrict__ Cf, bf16* __restrict__ Cb, int ldc, long cBatch,
    const float* __restrict__ bias, const float* __restrict__ resid, int act) {
    __shared__ __align__(16) bf16 AsB[2][128 * 32];
    __shared__ __align__(16) bf16 BtB[2][128 * 32];
    int z = blockIdx.z;
    A += (long)z * aBatch;
    long coff = (long)z * cBatch;
    int m0 = blockIdx.x * 128;
    int n0 = blockIdx.y * 128;
    int tid = threadIdx.x, lane = tid & 31, w = tid >> 5;
    int wm = w & 3, wn = w >> 2;

    unsigned ldsA[2] = {lds_off(&AsB[0][0]), lds_off(&AsB[1][0])};
    unsigned ldsBt[2] = {lds_off(&BtB[0][0]), lds_off(&BtB[1][0])};

    auto issue_tile = [&](int buf, int k0) {
#pragma unroll
        for (int u = 0; u < 2; ++u) {
            int i = tid + u * 256;
            int r = i >> 2, q = i & 3;
            async_b128(ldsA[buf] + r * 64 + q * 16,
                       (const char*)A + ((long)(m0 + r) * lda + k0) * 2 + q * 16);
        }
#pragma unroll
        for (int u = 0; u < 2; ++u) {
            int i = tid + u * 256;
            int r = i >> 2, q = i & 3;
            async_b128(ldsBt[buf] + r * 64 + q * 16,
                       (const char*)BwT + ((long)(n0 + r) * ldbT + k0) * 2 + q * 16);
        }
    };

    v8f acc[2][4];
#pragma unroll
    for (int i2 = 0; i2 < 2; ++i2)
#pragma unroll
        for (int j = 0; j < 4; ++j) acc[i2][j] = (v8f){0.f, 0.f, 0.f, 0.f, 0.f, 0.f, 0.f, 0.f};

    int nk = K >> 5;
    issue_tile(0, 0);
    for (int kt = 0; kt < nk; ++kt) {
        wait_async0();
        __syncthreads();
        if (kt + 1 < nk) issue_tile((kt + 1) & 1, (kt + 1) * 32);
        const bf16* Asb = AsB[kt & 1];
        const bf16* Btb = BtB[kt & 1];
        v16bf a0 = load_afrag_rm(Asb, 32, wm * 32, 0, lane);
        v16bf a1 = load_afrag_rm(Asb, 32, wm * 32 + 16, 0, lane);
#pragma unroll
        for (int j = 0; j < 4; ++j) {
            v16bf bf_ = load_bfrag_nm(Btb, 32, wn * 64 + j * 16, 0, lane);
            acc[0][j] = wmma_bf16(a0, bf_, acc[0][j]);
            acc[1][j] = wmma_bf16(a1, bf_, acc[1][j]);
        }
        __syncthreads();
    }

    int rb = m0 + wm * 32 + ((lane < 16) ? 0 : 8);
    int col_l = lane & 15;
#pragma unroll
    for (int i2 = 0; i2 < 2; ++i2) {
#pragma unroll
        for (int j = 0; j < 4; ++j) {
            int col = n0 + wn * 64 + j * 16 + col_l;
            float bs = bias ? bias[col] : 0.0f;
#pragma unroll
            for (int r = 0; r < 8; ++r) {
                int row = rb + i2 * 16 + r;
                float vv = acc[i2][j][r] + bs;
                if (act == 1) vv = gelu_exact(vv);
                long idx = (long)row * ldc + col + coff;
                if (resid) vv += resid[idx];
                if (Cf) Cf[idx] = vv;
                if (Cb) Cb[idx] = f2bf(vv);
            }
        }
    }
}

// ---------------- FAVOR+ feature post-processing ----------------
__device__ __forceinline__ void atomicMaxF(float* addr, float val) {
    int* ai = (int*)addr;
    int old = *ai;
    while (__int_as_float(old) < val) {
        int assumed = old;
        old = atomicCAS(ai, assumed, __float_as_int(val));
        if (old == assumed) break;
    }
}

__global__ __launch_bounds__(256) void k_init_kmax(float* kmax) { kmax[0] = -3e38f; }

__global__ __launch_bounds__(256) void k_kmax(const float* __restrict__ xd, float* __restrict__ kmax) {
    __shared__ float red[256];
    int r = blockIdx.x, h = blockIdx.y, tid = threadIdx.x;
    float v = xd[(long)r * HM + h * MFF + tid] * DN_SCALE;
    red[tid] = v;
    __syncthreads();
    for (int st = 128; st > 0; st >>= 1) {
        if (tid < st) red[tid] = fmaxf(red[tid], red[tid + st]);
        __syncthreads();
    }
    if (tid == 0) atomicMaxF(kmax, red[0]);
}

__global__ __launch_bounds__(256) void k_features(const float* __restrict__ xd, const float* __restrict__ qk,
                                                  const float* __restrict__ kmaxp, int isq,
                                                  bf16* __restrict__ out) {
    __shared__ float red[256];
    int r = blockIdx.x, h = blockIdx.y, tid = threadIdx.x;
    float val = xd[(long)r * HM + h * MFF + tid] * DN_SCALE;
    float qv = 0.0f;
    if (tid < DHH) {
        float t = qk[(long)r * EE + h * DHH + tid] * DN_SCALE;
        qv = t * t;
    }
    red[tid] = qv;
    __syncthreads();
    for (int st = 128; st > 0; st >>= 1) {
        if (tid < st) red[tid] += red[tid + st];
        __syncthreads();
    }
    float diag = 0.5f * red[0];
    __syncthreads();
    float mx;
    if (isq) {
        red[tid] = val;
        __syncthreads();
        for (int st = 128; st > 0; st >>= 1) {
            if (tid < st) red[tid] = fmaxf(red[tid], red[tid + st]);
            __syncthreads();
        }
        mx = red[0];
    } else {
        mx = kmaxp[0];
    }
    out[(long)r * HM + h * MFF + tid] = f2bf(RATIO * (expf(val - diag - mx) + 1e-4f));
}

// ---------------- causal linear attention: 3-pass parallel scan (all WMMA) ----------------
// Pass 1: per-chunk local state S_c = K_c^T @ Vaug [256 x 80] fp32 -> Spart[bh][c].
__global__ __launch_bounds__(256) void k_attn_part(const bf16* __restrict__ kf, const bf16* __restrict__ vb,
                                                   float* __restrict__ Spart) {
    extern __shared__ char smem[];
    bf16* Kc  = (bf16*)smem;               // 128*256
    bf16* VcT = Kc + 128 * 256;            // 80*128 (n-major)
    int c = blockIdx.x, bh = blockIdx.y;
    int b = bh >> 3, h = bh & 7;
    int c0 = c * 128;
    int tid = threadIdx.x, lane = tid & 31, w = tid >> 5;
    int col_l = lane & 15;
    long rowbase = (long)b * SS;
    unsigned ldsK = lds_off(Kc);

#if HAVE_TDM
    if (w == 0)   // single TDM descriptor moves the whole 128x256 bf16 tile
        tdm_load_2d(ldsK, (const char*)kf + ((rowbase + c0) * (long)HM + h * MFF) * 2,
                    HM, ROWS, MFF, 128, HM);
#else
    for (int i = tid; i < 4096; i += 256) {
        int t = i >> 5, q = i & 31;
        long gb = ((rowbase + c0 + t) * (long)HM + h * MFF) * 2 + q * 16;
        async_b128(ldsK + t * 512 + q * 16, (const char*)kf + gb);
    }
#endif
    bf16 bfone = f2bf(1.0f), bfzero = f2bf(0.0f);
    for (int i = tid; i < DA * 128; i += 256) {
        int d = i >> 7, t = i & 127;
        bf16 val;
        if (d < DHH)      val = vb[(rowbase + c0 + t) * (long)EE + h * DHH + d];
        else if (d == 64) val = bfone;
        else              val = bfzero;
        VcT[d * 128 + t] = val;
    }
#if HAVE_TDM
    wait_tensor0();
#else
    wait_async0();
#endif
    __syncthreads();

    float* outp = Spart + ((long)bh * NCH + c) * 256 * DA;
    for (int mi = 2 * w; mi < 2 * w + 2; ++mi) {
        int mrow = mi * 16 + ((lane < 16) ? 0 : 8);
#pragma unroll
        for (int nj = 0; nj < 5; ++nj) {
            v8f accS = (v8f){0.f, 0.f, 0.f, 0.f, 0.f, 0.f, 0.f, 0.f};
#pragma unroll
            for (int ks = 0; ks < 4; ++ks) {
                v16bf af  = load_afrag_cm(Kc, 256, mi * 16, ks * 32, lane);  // A[m][t]=Kc[t][m]
                v16bf bfr = load_bfrag_nm(VcT, 128, nj * 16, ks * 32, lane);
                accS = wmma_bf16(af, bfr, accS);
            }
#pragma unroll
            for (int r = 0; r < 8; ++r)
                outp[(long)(mrow + r) * DA + nj * 16 + col_l] = accS[r];
        }
    }
}

// Pass 2: exclusive fp32 prefix over chunks; emit bf16 in n-major [d][m] layout for pass 3.
__global__ __launch_bounds__(256) void k_attn_scan(const float* __restrict__ Spart,
                                                   bf16* __restrict__ SpreB) {
    int bh = blockIdx.y;
    int e = blockIdx.x * 256 + threadIdx.x;    // e in [0, 20480)
    int m = e / DA, d = e - m * DA;
    float run = 0.0f;
    for (int c = 0; c < NCH; ++c) {
        long pi = (((long)bh * NCH + c) * 256 + m) * DA + d;
        SpreB[(((long)bh * NCH + c) * DA + d) * 256 + m] = f2bf(run);
        run += Spart[pi];
    }
}

// Pass 3: D = Qc @ S_prefix + tril(Qc Kc^T) @ Vaug ; out = D[:,:64]/(D[:,64]+1e-6).
__global__ __launch_bounds__(256) void k_attn_out(const bf16* __restrict__ qf, const bf16* __restrict__ kf,
                                                  const bf16* __restrict__ vb, const bf16* __restrict__ SpreB,
                                                  bf16* __restrict__ ob) {
    extern __shared__ char smem[];
    bf16* Qc  = (bf16*)smem;               // 128*256
    bf16* Kc  = Qc + 128 * 256;            // 128*256
    bf16* Sb  = Kc + 128 * 256;            // 80*256 (n-major [d][m])
    bf16* VcT = Sb + DA * 256;             // 80*128
    bf16* Ac  = VcT + DA * 128;            // 128*128
    int c = blockIdx.x, bh = blockIdx.y;
    int b = bh >> 3, h = bh & 7;
    int c0 = c * 128;
    int tid = threadIdx.x, lane = tid & 31, w = tid >> 5;
    int col_l = lane & 15;
    long rowbase = (long)b * SS;
    unsigned ldsQ = lds_off(Qc), ldsK = lds_off(Kc), ldsS = lds_off(Sb);
    const char* sgb = (const char*)(SpreB + ((long)bh * NCH + c) * DA * 256);

#if HAVE_TDM
    if (w == 0)
        tdm_load_2d(ldsQ, (const char*)qf + ((rowbase + c0) * (long)HM + h * MFF) * 2,
                    HM, ROWS, MFF, 128, HM);
    else if (w == 1)
        tdm_load_2d(ldsK, (const char*)kf + ((rowbase + c0) * (long)HM + h * MFF) * 2,
                    HM, ROWS, MFF, 128, HM);
    else if (w == 2)
        tdm_load_2d(ldsS, sgb, 256, DA, 256, DA, 256);
#else
    for (int i = tid; i < 4096; i += 256) {
        int t = i >> 5, q = i & 31;
        long gb = ((rowbase + c0 + t) * (long)HM + h * MFF) * 2 + q * 16;
        async_b128(ldsQ + t * 512 + q * 16, (const char*)qf + gb);
        async_b128(ldsK + t * 512 + q * 16, (const char*)kf + gb);
    }
    for (int i = tid; i < 2560; i += 256) {
        async_b128(ldsS + i * 16, sgb + i * 16);
    }
#endif
    bf16 bfone = f2bf(1.0f), bfzero = f2bf(0.0f);
    for (int i = tid; i < DA * 128; i += 256) {
        int d = i >> 7, t = i & 127;
        bf16 val;
        if (d < DHH)      val = vb[(rowbase + c0 + t) * (long)EE + h * DHH + d];
        else if (d == 64) val = bfone;
        else              val = bfzero;
        VcT[d * 128 + t] = val;
    }
#if HAVE_TDM
    wait_tensor0();
#else
    wait_async0();
#endif
    __syncthreads();

    // Phase A: A = tril(Qc @ Kc^T), wave w owns row tile w
    for (int tj = 0; tj < 8; ++tj) {
        if (tj <= w) {
            v8f accA = (v8f){0.f, 0.f, 0.f, 0.f, 0.f, 0.f, 0.f, 0.f};
#pragma unroll
            for (int ks = 0; ks < 8; ++ks) {
                v16bf af  = load_afrag_rm(Qc, 256, w * 16, ks * 32, lane);
                v16bf bfr = load_bfrag_nm(Kc, 256, tj * 16, ks * 32, lane);
                accA = wmma_bf16(af, bfr, accA);
            }
#pragma unroll
            for (int r = 0; r < 8; ++r) {
                int tt = w * 16 + r + ((lane < 16) ? 0 : 8);
                int tp = tj * 16 + col_l;
                Ac[tt * 128 + tp] = (tp <= tt) ? f2bf(accA[r]) : bfzero;
            }
        } else {
#pragma unroll
            for (int r = 0; r < 8; ++r) {
                int tt = w * 16 + r + ((lane < 16) ? 0 : 8);
                Ac[tt * 128 + tj * 16 + col_l] = bfzero;
            }
        }
    }
    __syncthreads();

    // Phase B: D = Qc @ S_prefix + Ac @ Vaug
    v8f accD[5];
#pragma unroll
    for (int nj = 0; nj < 5; ++nj) accD[nj] = (v8f){0.f, 0.f, 0.f, 0.f, 0.f, 0.f, 0.f, 0.f};
    for (int ks = 0; ks < 8; ++ks) {
        v16bf af = load_afrag_rm(Qc, 256, w * 16, ks * 32, lane);
#pragma unroll
        for (int nj = 0; nj < 5; ++nj) {
            v16bf bfr = load_bfrag_nm(Sb, 256, nj * 16, ks * 32, lane);
            accD[nj] = wmma_bf16(af, bfr, accD[nj]);
        }
    }
    for (int ks = 0; ks < 4; ++ks) {
        v16bf af = load_afrag_rm(Ac, 128, w * 16, ks * 32, lane);
#pragma unroll
        for (int nj = 0; nj < 5; ++nj) {
            v16bf bfr = load_bfrag_nm(VcT, 128, nj * 16, ks * 32, lane);
            accD[nj] = wmma_bf16(af, bfr, accD[nj]);
        }
    }
#pragma unroll
    for (int r = 0; r < 8; ++r) {
        float den = __shfl(accD[4][r], (lane < 16) ? 0 : 16, 32);
        float inv = 1.0f / (den + 1e-6f);
        int tt = w * 16 + r + ((lane < 16) ? 0 : 8);
        long orow = (rowbase + c0 + tt) * (long)EE + h * DHH;
#pragma unroll
        for (int nj = 0; nj < 4; ++nj)
            ob[orow + nj * 16 + col_l] = f2bf(accD[nj][r] * inv);
    }
}

// ---------------- host side ----------------
static inline size_t alignup(size_t x) { return (x + 255) & ~(size_t)255; }

extern "C" void kernel_launch(void* const* d_in, const int* in_sizes, int n_in,
                              void* d_out, int out_size, void* d_ws, size_t ws_size,
                              hipStream_t stream) {
    const int*   src      = (const int*)d_in[0];
    const float* emb      = (const float*)d_in[2];
    const float* ln_emb_g = (const float*)d_in[3];
    const float* ln_emb_b = (const float*)d_in[4];
    const float* Wq       = (const float*)d_in[5];
    const float* Wk       = (const float*)d_in[6];
    const float* Wv       = (const float*)d_in[7];
    const float* Wo       = (const float*)d_in[8];
    const float* ln1_g    = (const float*)d_in[9];
    const float* ln1_b    = (const float*)d_in[10];
    const float* W1       = (const float*)d_in[11];
    const float* b1       = (const float*)d_in[12];
    const float* W2       = (const float*)d_in[13];
    const float* b2       = (const float*)d_in[14];
    const float* ln2_g    = (const float*)d_in[15];
    const float* ln2_b    = (const float*)d_in[16];
    const float* proj     = (const float*)d_in[17];
    const float* fc_w     = (const float*)d_in[18];
    const float* fc_b     = (const float*)d_in[19];
    float* out = (float*)d_out;

    char* base = (char*)d_ws;
    size_t off = 0;
    auto take = [&](size_t bytes) -> char* {
        char* r = base + off;
        off = alignup(off + bytes);
        return r;
    };
    bf16* wqT   = (bf16*)take((size_t)LL * EE * EE * 2);
    bf16* wkT   = (bf16*)take((size_t)LL * EE * EE * 2);
    bf16* wvT   = (bf16*)take((size_t)LL * EE * EE * 2);
    bf16* woT   = (bf16*)take((size_t)LL * EE * EE * 2);
    bf16* w1T   = (bf16*)take((size_t)LL * EE * 4 * EE * 2);
    bf16* w2T   = (bf16*)take((size_t)LL * 4 * EE * EE * 2);
    bf16* projB = (bf16*)take((size_t)LL * MFF * DHH * 2);
    bf16* fcwT  = (bf16*)take((size_t)EE * VV * 2);
    float* tmp  = (float*)take((size_t)ROWS * EE * 4);
    float* x    = (float*)take((size_t)ROWS * EE * 4);
    bf16* hb    = (bf16*)take((size_t)ROWS * EE * 2);
    float* q32  = (float*)take((size_t)ROWS * EE * 4);
    float* k32  = (float*)take((size_t)ROWS * EE * 4);
    bf16* qb    = (bf16*)take((size_t)ROWS * EE * 2);
    bf16* kb    = (bf16*)take((size_t)ROWS * EE * 2);
    bf16* vbuf  = (bf16*)take((size_t)ROWS * EE * 2);
    float* xd   = (float*)take((size_t)ROWS * HM * 4);
    bf16* qfb   = (bf16*)take((size_t)ROWS * HM * 2);
    bf16* kfb   = (bf16*)take((size_t)ROWS * HM * 2);
    bf16* obuf  = (bf16*)take((size_t)ROWS * EE * 2);
    bf16* ffb   = (bf16*)take((size_t)ROWS * 4 * EE * 2);
    bf16* xb    = (bf16*)take((size_t)ROWS * EE * 2);
    float* Spart = (float*)take((size_t)BB * HH * NCH * 256 * DA * 4);
    bf16*  SpreB = (bf16*)take((size_t)BB * HH * NCH * DA * 256 * 2);
    float* kmax = (float*)take(256);
    (void)ws_size; (void)in_sizes; (void)n_in; (void)out_size;

    auto cvt = [&](const float* in, bf16* o, long n) {
        int blocks = (int)((n + 255) / 256);
        if (blocks > 8192) blocks = 8192;
        k_f2b<<<blocks, 256, 0, stream>>>(in, o, n);
    };
    auto cvtT = [&](const float* in, bf16* o, int K, int N, int zB) {
        k_f2bT<<<dim3(K / 64, N / 64, zB), 256, 0, stream>>>(in, o, K, N);
    };
    cvtT(Wq, wqT, EE, EE, LL);
    cvtT(Wk, wkT, EE, EE, LL);
    cvtT(Wv, wvT, EE, EE, LL);
    cvtT(Wo, woT, EE, EE, LL);
    cvtT(W1, w1T, EE, 4 * EE, LL);
    cvtT(W2, w2T, 4 * EE, EE, LL);
    cvtT(fc_w, fcwT, EE, VV, 1);
    cvt(proj, projB, (long)LL * MFF * DHH);

    k_embed<<<ROWS, 256, 0, stream>>>(src, emb, tmp);
    k_ln<<<ROWS, 256, 0, stream>>>(tmp, ln_emb_g, ln_emb_b, x, nullptr);

    auto gemm = [&](const bf16* A, int lda, long aB, const bf16* BT, int N, int K,
                    float* Cf, bf16* Cb, int ldc, long cB, const float* bias, const float* resid,
                    int act, int gz) {
        dim3 grid(ROWS / 128, N / 128, gz);
        k_gemm<<<grid, 256, 0, stream>>>(A, lda, aB, BT, K, ROWS, N, K, Cf, Cb, ldc, cB,
                                         bias, resid, act);
    };

    const size_t PART_LDS = 128 * 256 * 2 + DA * 128 * 2;
    const size_t OUT_LDS  = 2 * (128 * 256 * 2) + DA * 256 * 2 + DA * 128 * 2 + 128 * 128 * 2;

    for (int l = 0; l < LL; ++l) {
        const bf16* wq = wqT + (size_t)l * EE * EE;
        const bf16* wk = wkT + (size_t)l * EE * EE;
        const bf16* wv = wvT + (size_t)l * EE * EE;
        const bf16* wo = woT + (size_t)l * EE * EE;
        const bf16* w1 = w1T + (size_t)l * EE * 4 * EE;
        const bf16* w2 = w2T + (size_t)l * 4 * EE * EE;
        const bf16* pt = projB + (size_t)l * MFF * DHH;

        k_ln<<<ROWS, 256, 0, stream>>>(x, ln1_g + l * EE, ln1_b + l * EE, nullptr, hb);
        gemm(hb, EE, 0, wq, EE, EE, q32, qb, EE, 0, nullptr, nullptr, 0, 1);
        gemm(hb, EE, 0, wk, EE, EE, k32, kb, EE, 0, nullptr, nullptr, 0, 1);
        gemm(hb, EE, 0, wv, EE, EE, nullptr, vbuf, EE, 0, nullptr, nullptr, 0, 1);

        gemm(qb, EE, DHH, pt, MFF, DHH, xd, nullptr, HM, MFF, nullptr, nullptr, 0, HH);
        k_features<<<dim3(ROWS, HH), 256, 0, stream>>>(xd, q32, nullptr, 1, qfb);

        gemm(kb, EE, DHH, pt, MFF, DHH, xd, nullptr, HM, MFF, nullptr, nullptr, 0, HH);
        k_init_kmax<<<1, 1, 0, stream>>>(kmax);
        k_kmax<<<dim3(ROWS, HH), 256, 0, stream>>>(xd, kmax);
        k_features<<<dim3(ROWS, HH), 256, 0, stream>>>(xd, k32, kmax, 0, kfb);

        k_attn_part<<<dim3(NCH, BB * HH), 256, PART_LDS, stream>>>(kfb, vbuf, Spart);
        k_attn_scan<<<dim3(80, BB * HH), 256, 0, stream>>>(Spart, SpreB);
        k_attn_out<<<dim3(NCH, BB * HH), 256, OUT_LDS, stream>>>(qfb, kfb, vbuf, SpreB, obuf);

        gemm(obuf, EE, 0, wo, EE, EE, x, nullptr, EE, 0, nullptr, x, 0, 1);

        k_ln<<<ROWS, 256, 0, stream>>>(x, ln2_g + l * EE, ln2_b + l * EE, nullptr, hb);
        gemm(hb, EE, 0, w1, 4 * EE, EE, nullptr, ffb, 4 * EE, 0, b1 + l * 4 * EE, nullptr, 1, 1);
        gemm(ffb, 4 * EE, 0, w2, EE, 4 * EE, x, nullptr, EE, 0, b2 + l * EE, x, 0, 1);
    }

    cvt(x, xb, (long)ROWS * EE);
    gemm(xb, EE, 0, fcwT, VV, EE, out, nullptr, VV, 0, fc_b, nullptr, 0, 1);
}
